// StructureModule_41532333753165
// MI455X (gfx1250) — matrix operand order repaired and checked by reading, not compiled, not run on hardware
//
#include <hip/hip_runtime.h>
#include <hip/hip_bf16.h>
#include <stdint.h>

typedef __attribute__((ext_vector_type(16))) __bf16 v16bf;
typedef __attribute__((ext_vector_type(8)))  float  v8f;
typedef unsigned int uint32x4 __attribute__((ext_vector_type(4)));
typedef int          int32x8  __attribute__((ext_vector_type(8)));
typedef int          int32x4  __attribute__((ext_vector_type(4)));

#define NTOK   768
#define CS     384
#define CZ     128
#define CH     16
#define NH     12
#define PQN    4
#define PVN    8
#define CCAT   2112
#define INFV   100000.0f

#define BM 64
#define BN 32
#define BK 64
#define LDSP 68   // padded row length (floats): 272B rows = 16B aligned, spreads banks

#if __has_builtin(__builtin_amdgcn_tensor_load_to_lds) && __has_builtin(__builtin_amdgcn_s_wait_tensorcnt)
#define HAVE_TDM 1
#endif

__device__ __forceinline__ __bf16 to_bf(float x) { return (__bf16)x; }

// ---------------------------------------------------------------------------
// Batched block GEMM: C[b] = act(A[b] @ B[b] + bias) (+ resid[b])
// 256 threads = 8 waves; block computes BM x BN of C; each wave one 16x16 WMMA
// tile, K consumed in BK=64 chunks (2 x v_wmma_f32_16x16x32_bf16 per chunk).
// A tile staged by the Tensor Data Mover (tensor_load_to_lds, TENSORcnt) when
// the builtin is available; fallback is coalesced float4 loads + prefetch.
// B staged transposed so WMMA operand reads from LDS are contiguous.
// K must be a multiple of 64.  act: 0=none 1=relu 2=sigmoid
// ---------------------------------------------------------------------------
__global__ __launch_bounds__(256)
void gemm_bf16_wmma(const float* __restrict__ A, int ldA, long long sA,
                    const float* __restrict__ Bm, int ldB, long long sB,
                    const float* __restrict__ bias,
                    const float* __restrict__ resid, int ldR, long long sR,
                    float* __restrict__ C, int ldC, long long sC,
                    int M, int N, int K, int act)
{
  __shared__ float As[BM][LDSP];   // [m][k]  (row stride 68 floats = 64 + 4 pad)
  __shared__ float Bs[BN][LDSP];   // [n][k]  (transposed stage)

  const int tid = threadIdx.x;
  const int b   = blockIdx.z;
  A  += (long long)b * sA;
  Bm += (long long)b * sB;
  C  += (long long)b * sC;
  if (resid) resid += (long long)b * sR;

  const int m0 = blockIdx.y * BM;
  const int n0 = blockIdx.x * BN;

  const int wave = tid >> 5;
  const int lane = tid & 31;
  const int half = lane >> 4;
  const int l15  = lane & 15;
  const int wm   = wave & 3;   // 4 waves along M
  const int wn   = wave >> 2;  // 2 waves along N

  v8f acc = {0.f,0.f,0.f,0.f,0.f,0.f,0.f,0.f};

  for (int k0 = 0; k0 < K; k0 += BK) {
#ifdef HAVE_TDM
    // ---- stage A (BM x BK) via Tensor Data Mover ----
    // D# group0: count=1 | lds_addr | global_addr(57b) | type=2
    // D# group1: data_size=4B, pad_enable, pad_interval=64DW, pad_amount=4DW
    //            (matches LDSP=68-float rows), tensor_dim0=K-k0 (no k OOB),
    //            tensor_dim1=M-m0 (OOB rows zero-filled), tile 64x64,
    //            dim0 stride = ldA.
    if (wave == 0) {
      const uint64_t ga = (uint64_t)(uintptr_t)(A + (long long)m0 * ldA + k0);
      uint32x4 g0;
      g0[0] = 1u;
      g0[1] = (unsigned)(uintptr_t)&As[0][0];
      g0[2] = (unsigned)ga;
      g0[3] = (unsigned)(ga >> 32) | (2u << 30);
      const unsigned dim0 = (unsigned)(K - k0);
      const unsigned dim1 = (unsigned)(M - m0);
      int32x8 g1;
      g1[0] = (int)((2u << 16) | (1u << 20) | (5u << 22) | (3u << 25));
      g1[1] = (int)((dim0 & 0xFFFFu) << 16);
      g1[2] = (int)(((dim0 >> 16) & 0xFFFFu) | ((dim1 & 0xFFFFu) << 16));
      g1[3] = (int)(((dim1 >> 16) & 0xFFFFu) | ((unsigned)BK << 16));
      g1[4] = (int)BM;                 // tile_dim1; tile_dim2 = 0 (2D)
      g1[5] = (int)(unsigned)ldA;      // tensor_dim0_stride[31:0]
      g1[6] = 0;
      g1[7] = 0;
      const int32x4 gz = {0, 0, 0, 0};
#if defined(__clang_major__) && (__clang_major__ >= 23)
      const int32x8 gz8 = {0, 0, 0, 0, 0, 0, 0, 0};
      __builtin_amdgcn_tensor_load_to_lds(g0, g1, gz, gz, gz8, 0);
#else
      __builtin_amdgcn_tensor_load_to_lds(g0, g1, gz, gz, 0);
#endif
      __builtin_amdgcn_s_wait_tensorcnt(0);
    }
#else
    // ---- stage A (BM x BK): coalesced float4 loads ----
    {
      const int ar = tid >> 2;
      const int ak = (tid & 3) * 16;
      const int gm = m0 + ar;
      float4 v0, v1, v2, v3;
      if (gm < M) {
        const float* src = A + (long long)gm * ldA + k0 + ak;
        v0 = *(const float4*)(src + 0);
        v1 = *(const float4*)(src + 4);
        v2 = *(const float4*)(src + 8);
        v3 = *(const float4*)(src + 12);
      } else {
        v0 = v1 = v2 = v3 = make_float4(0.f, 0.f, 0.f, 0.f);
      }
      *(float4*)&As[ar][ak + 0]  = v0;
      *(float4*)&As[ar][ak + 4]  = v1;
      *(float4*)&As[ar][ak + 8]  = v2;
      *(float4*)&As[ar][ak + 12] = v3;
      if (k0 + BK < K && gm < M)
        __builtin_prefetch(A + (long long)gm * ldA + k0 + BK + ak, 0, 0);
    }
#endif
    // ---- stage B (BK x BN) transposed into Bs[n][k] ----
#pragma unroll
    for (int i = 0; i < (BK * BN) / 256; ++i) {
      const int e  = tid + i * 256;
      const int kk = e >> 5;        // 0..63
      const int n  = e & 31;        // 0..31
      const int gn = n0 + n;
      float bval = 0.f;
      if (gn < N) bval = Bm[(long long)(k0 + kk) * ldB + gn];
      Bs[n][kk] = bval;
    }
    __syncthreads();

    // ---- 2 WMMA steps over the staged BK ----
#pragma unroll
    for (int ks = 0; ks < BK; ks += 32) {
      v16bf av, bv;
      const int kA0 = ks + 8 * half;
      const float* arow = &As[wm * 16 + l15][0];
#pragma unroll
      for (int j2 = 0; j2 < 4; ++j2) {
        av[2*j2]         = to_bf(arow[kA0 + 2*j2]);
        av[2*j2 + 1]     = to_bf(arow[kA0 + 2*j2 + 1]);
        av[8 + 2*j2]     = to_bf(arow[kA0 + 16 + 2*j2]);
        av[8 + 2*j2 + 1] = to_bf(arow[kA0 + 16 + 2*j2 + 1]);
      }
      const int kB0 = ks + 8 * half;
      const float* brow = &Bs[wn * 16 + l15][0];
#pragma unroll
      for (int e = 0; e < 16; ++e) bv[e] = to_bf(brow[kB0 + (e & 7) + 16 * (e >> 3)]);
      acc = __builtin_amdgcn_wmma_f32_16x16x32_bf16(false, av, false, bv,
                                                    (short)0, acc, false, false);
    }
    __syncthreads();
  }

  // ---- epilogue ----
#pragma unroll
  for (int r = 0; r < 8; ++r) {
    const int mi = m0 + wm * 16 + r + 8 * half;
    const int ni = n0 + wn * 16 + l15;
    if (mi < M && ni < N) {
      float v = acc[r];
      if (bias) v += bias[ni];
      if (act == 1)      v = fmaxf(v, 0.f);
      else if (act == 2) v = 1.f / (1.f + expf(-v));
      if (resid) v += resid[(long long)mi * ldR + ni];
      C[(long long)mi * ldC + ni] = v;
    }
  }
}

// ---------------------------------------------------------------------------
// Attention logits + softmax for one (head, 16-row i-tile).
// Q.K^T via WMMA (C_H=16 zero-padded to K=32), adds pair bias, point term,
// mask; row softmax over j=768; writes probs a[i][h][j] fp32.
// ---------------------------------------------------------------------------
__global__ __launch_bounds__(256)
void attn_scores_softmax(const float* __restrict__ q,       // [768][12][16]
                         const float* __restrict__ kv,      // [768][12][32] (k|v)
                         const float* __restrict__ q_pts,   // [768][12][4][3]
                         const float* __restrict__ k_pts,   // [768][12][4][3]
                         const float* __restrict__ bias_z,  // [(i*768+j)][12]
                         const float* __restrict__ mask,    // [768]
                         const float* __restrict__ head_w,  // [12]
                         float* __restrict__ a_out)         // [768][12][768]
{
  __shared__ float sc[16 * NTOK];
  __shared__ float qs[16 * 12];
  __shared__ float red[256];

  const int it = blockIdx.x;
  const int h  = blockIdx.y;
  const int tid  = threadIdx.x;
  const int lane = tid & 31;
  const int wv   = tid >> 5;
  const int half = lane >> 4;
  const int l15  = lane & 15;

  // ---- Q tile (16x32, c>=16 zero) ----
  const int irow = it * 16 + l15;
  v16bf av;
#pragma unroll
  for (int j2 = 0; j2 < 4; ++j2) {
    const int c0 = 8 * half + 2 * j2;            // < 16 : valid
    av[2*j2]     = to_bf(q[irow * (NH*CH) + h * CH + c0]);
    av[2*j2 + 1] = to_bf(q[irow * (NH*CH) + h * CH + c0 + 1]);
    av[8 + 2*j2]     = to_bf(0.f);
    av[8 + 2*j2 + 1] = to_bf(0.f);
  }

  const float qk_scale = 0.14433756729740643f;   // 1/sqrt(3*C_H)

  for (int jt = wv; jt < NTOK / 16; jt += 8) {
    const int jcol = jt * 16 + l15;
    v16bf bv;
#pragma unroll
    for (int e = 0; e < 16; ++e) {
      const int c = 8 * half + (e & 7) + 16 * (e >> 3); // e>=8 -> pad zeros
      float kval = 0.f;
      if (c < CH) kval = kv[jcol * (NH * 2 * CH) + h * (2 * CH) + c];
      bv[e] = to_bf(kval);
    }
    v8f acc = {0.f,0.f,0.f,0.f,0.f,0.f,0.f,0.f};
    acc = __builtin_amdgcn_wmma_f32_16x16x32_bf16(false, av, false, bv,
                                                  (short)0, acc, false, false);
#pragma unroll
    for (int r = 0; r < 8; ++r)
      sc[(r + 8 * half) * NTOK + jt * 16 + l15] = acc[r] * qk_scale;
  }

  if (tid < 16 * 12)
    qs[tid] = q_pts[(long long)(it * 16 + (tid / 12)) * (NH * PQN * 3) + h * 12 + (tid % 12)];
  __syncthreads();

  const float hw = log1pf(expf(head_w[h])) * 0.13608276348795434f; // softplus*sqrt(1/54)
  const float bscale = 0.5773502691896258f;                        // sqrt(1/3)
  for (int e = tid; e < 16 * NTOK; e += 256) {
    const int il = e / NTOK;
    const int j  = e - il * NTOK;
    const long long i = (long long)(it * 16 + il);
    float pts = 0.f;
#pragma unroll
    for (int t = 0; t < 12; ++t) {
      const float d = qs[il * 12 + t] - k_pts[(long long)j * (NH * PQN * 3) + h * 12 + t];
      pts += d * d;
    }
    float v = sc[e];
    v += bscale * bias_z[(i * NTOK + j) * NH + h];
    v += -0.5f * hw * pts;
    v += INFV * (mask[j] - 1.f);
    sc[e] = v;
  }
  __syncthreads();

  // ---- softmax over j (16 threads per row) ----
  const int row = tid >> 4;
  const int sub = tid & 15;
  float lmax = -1e30f;
  for (int kk = 0; kk < NTOK / 16; ++kk)
    lmax = fmaxf(lmax, sc[row * NTOK + sub + 16 * kk]);
  red[tid] = lmax;
  __syncthreads();
  float rmax = -1e30f;
#pragma unroll
  for (int u = 0; u < 16; ++u) rmax = fmaxf(rmax, red[row * 16 + u]);
  __syncthreads();
  float lsum = 0.f;
  for (int kk = 0; kk < NTOK / 16; ++kk) {
    const int idx = row * NTOK + sub + 16 * kk;
    const float ev = expf(sc[idx] - rmax);
    sc[idx] = ev;
    lsum += ev;
  }
  red[tid] = lsum;
  __syncthreads();
  float rsum = 0.f;
#pragma unroll
  for (int u = 0; u < 16; ++u) rsum += red[row * 16 + u];
  const float inv = 1.f / rsum;
  const long long i = (long long)(it * 16 + row);
  for (int kk = 0; kk < NTOK / 16; ++kk) {
    const int j = sub + 16 * kk;
    a_out[(i * NH + h) * NTOK + j] = sc[row * NTOK + j] * inv;
  }
}

// ---------------------------------------------------------------------------
// Rigid transforms for the point projections
// ---------------------------------------------------------------------------
__global__ void qp_transform(const float* __restrict__ raw,   // [768][144]
                             const float* __restrict__ rot,   // [768][3][3]
                             const float* __restrict__ trans, // [768][3]
                             float* __restrict__ q_pts)       // [768][12][4][3]
{
  const int tid = blockIdx.x * blockDim.x + threadIdx.x;
  if (tid >= NTOK * 48) return;
  const int n = tid / 48, p = tid - n * 48;
  const float lx = raw[n * 144 + p];
  const float ly = raw[n * 144 + 48 + p];
  const float lz = raw[n * 144 + 96 + p];
#pragma unroll
  for (int x = 0; x < 3; ++x) {
    q_pts[n * 144 + p * 3 + x] =
        rot[n * 9 + x * 3 + 0] * lx + rot[n * 9 + x * 3 + 1] * ly +
        rot[n * 9 + x * 3 + 2] * lz + trans[n * 3 + x];
  }
}

__global__ void kvp_transform(const float* __restrict__ raw,   // [768][432]
                              const float* __restrict__ rot,
                              const float* __restrict__ trans,
                              float* __restrict__ k_pts,       // [768][12][4][3]
                              float* __restrict__ v_pts)       // [768][12][8][3]
{
  const int tid = blockIdx.x * blockDim.x + threadIdx.x;
  if (tid >= NTOK * 144) return;
  const int n = tid / 144, p = tid - n * 144;
  const float lx = raw[n * 432 + p];
  const float ly = raw[n * 432 + 144 + p];
  const float lz = raw[n * 432 + 288 + p];
  const int hh = p / 12, pp = p - hh * 12;
#pragma unroll
  for (int x = 0; x < 3; ++x) {
    const float g = rot[n * 9 + x * 3 + 0] * lx + rot[n * 9 + x * 3 + 1] * ly +
                    rot[n * 9 + x * 3 + 2] * lz + trans[n * 3 + x];
    if (pp < PQN) k_pts[n * 144 + hh * 12 + pp * 3 + x] = g;
    else          v_pts[n * 288 + hh * 24 + (pp - PQN) * 3 + x] = g;
  }
}

// o_pt: inverse frame + norm, scatter into concat layout
__global__ void opt_finish(const float* __restrict__ optb,  // [768][12][8][3]
                           const float* __restrict__ rot,
                           const float* __restrict__ trans,
                           float* __restrict__ fo)           // [768][2112]
{
  const int tid = blockIdx.x * blockDim.x + threadIdx.x;
  if (tid >= NTOK * 96) return;
  const int n = tid / 96, idx = tid - n * 96;
  const float px = optb[n * 288 + idx * 3 + 0] - trans[n * 3 + 0];
  const float py = optb[n * 288 + idx * 3 + 1] - trans[n * 3 + 1];
  const float pz = optb[n * 288 + idx * 3 + 2] - trans[n * 3 + 2];
  float o[3];
#pragma unroll
  for (int x = 0; x < 3; ++x)   // R^T : sum_y rot[y][x] * v[y]
    o[x] = rot[n * 9 + 0 * 3 + x] * px + rot[n * 9 + 1 * 3 + x] * py +
           rot[n * 9 + 2 * 3 + x] * pz;
  fo[n * CCAT + 192 + idx] = o[0];
  fo[n * CCAT + 288 + idx] = o[1];
  fo[n * CCAT + 384 + idx] = o[2];
  fo[n * CCAT + 480 + idx] = sqrtf(o[0]*o[0] + o[1]*o[1] + o[2]*o[2] + 1e-8f);
}

__global__ void mul_inplace(float* __restrict__ a, const float* __restrict__ b, int n)
{
  const int tid = blockIdx.x * blockDim.x + threadIdx.x;
  if (tid < n) a[tid] *= b[tid];
}

__global__ __launch_bounds__(256)
void layernorm_row(const float* __restrict__ x, const float* __restrict__ g,
                   const float* __restrict__ b, float* __restrict__ y)
{
  __shared__ float red[256];
  const int n = blockIdx.x, t = threadIdx.x;
  float s = 0.f;
  for (int c = t; c < CS; c += 256) s += x[n * CS + c];
  red[t] = s; __syncthreads();
  for (int o = 128; o > 0; o >>= 1) { if (t < o) red[t] += red[t + o]; __syncthreads(); }
  const float mu = red[0] / (float)CS;
  __syncthreads();
  float v = 0.f;
  for (int c = t; c < CS; c += 256) { const float d = x[n * CS + c] - mu; v += d * d; }
  red[t] = v; __syncthreads();
  for (int o = 128; o > 0; o >>= 1) { if (t < o) red[t] += red[t + o]; __syncthreads(); }
  const float inv = rsqrtf(red[0] / (float)CS + 1e-5f);
  __syncthreads();
  for (int c = t; c < CS; c += 256)
    y[n * CS + c] = (x[n * CS + c] - mu) * inv * g[c] + b[c];
}

__global__ void backbone_update(const float* __restrict__ s, const float* __restrict__ w,
                                const float* __restrict__ b, float* __restrict__ out)
{
  const int tid = blockIdx.x * blockDim.x + threadIdx.x;
  if (tid >= NTOK * 3) return;
  const int n = tid / 3, d = tid - n * 3;
  float acc = b[d];
  for (int k = 0; k < CS; ++k) acc += s[n * CS + k] * w[k * 3 + d];
  out[tid] = acc;
}

// ---------------------------------------------------------------------------
static inline void launch_gemm(hipStream_t st,
                               const float* A, int ldA, long long sA,
                               const float* B, int ldB, long long sB,
                               const float* bias,
                               const float* resid, int ldR, long long sR,
                               float* C, int ldC, long long sC,
                               int M, int N, int K, int act, int batch)
{
  dim3 grid((N + BN - 1) / BN, (M + BM - 1) / BM, batch);
  gemm_bf16_wmma<<<grid, 256, 0, st>>>(A, ldA, sA, B, ldB, sB, bias,
                                       resid, ldR, sR, C, ldC, sC, M, N, K, act);
}

extern "C" void kernel_launch(void* const* d_in, const int* in_sizes, int n_in,
                              void* d_out, int out_size, void* d_ws, size_t ws_size,
                              hipStream_t stream)
{
  const float* s     = (const float*)d_in[0];
  const float* z     = (const float*)d_in[1];
  const float* rot   = (const float*)d_in[2];
  const float* trans = (const float*)d_in[3];
  const float* mask  = (const float*)d_in[4];
  const float* wq  = (const float*)d_in[7];  const float* bq  = (const float*)d_in[8];
  const float* wkv = (const float*)d_in[9];  const float* bkv = (const float*)d_in[10];
  const float* wqp = (const float*)d_in[11]; const float* bqp = (const float*)d_in[12];
  const float* wkvp= (const float*)d_in[13]; const float* bkvp= (const float*)d_in[14];
  const float* wb  = (const float*)d_in[15]; const float* bbv = (const float*)d_in[16];
  const float* hwp = (const float*)d_in[17];
  const float* wg  = (const float*)d_in[18]; const float* bg  = (const float*)d_in[19];
  const float* wo  = (const float*)d_in[20]; const float* bo  = (const float*)d_in[21];
  const float* ln1g= (const float*)d_in[22]; const float* ln1b= (const float*)d_in[23];
  const float* tw1 = (const float*)d_in[24]; const float* tb1 = (const float*)d_in[25];
  const float* tw2 = (const float*)d_in[26]; const float* tb2 = (const float*)d_in[27];
  const float* tw3 = (const float*)d_in[28]; const float* tb3 = (const float*)d_in[29];
  const float* ln2g= (const float*)d_in[30]; const float* ln2b= (const float*)d_in[31];
  const float* bbw = (const float*)d_in[32]; const float* bbb = (const float*)d_in[33];

  // ---- workspace carve-up (floats) ----
  float* p = (float*)d_ws;
  float* q_buf   = p; p += NTOK * 192;
  float* kv_buf  = p; p += NTOK * 384;
  float* qp_raw  = p; p += NTOK * 144;
  float* kvp_raw = p; p += NTOK * 432;
  float* q_pts   = p; p += NTOK * 144;
  float* k_pts   = p; p += NTOK * 144;
  float* v_pts   = p; p += NTOK * 288;
  float* bias_z  = p; p += (long long)NTOK * NTOK * NH;
  float* a_buf   = p; p += (long long)NTOK * NH * NTOK;
  float* fo      = p; p += NTOK * CCAT;
  float* g_buf   = p; p += NTOK * CCAT;
  float* optb    = p; p += NTOK * 288;
  float* s_ipa   = p; p += NTOK * CS;
  float* s_ln    = p; p += NTOK * CS;
  float* t1      = p; p += NTOK * CS;
  float* t2      = p; p += NTOK * CS;
  float* s2      = p; p += NTOK * CS;

  float* out_s = (float*)d_out;                       // [768][384]
  float* out_z = out_s + NTOK * CS;                   // passthrough z
  float* out_t = out_z + (long long)NTOK * NTOK * CZ; // [768][3]

  // z passthrough (independent of the compute chain)
  hipMemcpyAsync(out_z, z, (size_t)NTOK * NTOK * CZ * sizeof(float),
                 hipMemcpyDeviceToDevice, stream);

  // ---- s projections ----
  launch_gemm(stream, s, CS, 0, wq, 192, 0, bq, nullptr, 0, 0,
              q_buf, 192, 0, NTOK, 192, CS, 0, 1);
  launch_gemm(stream, s, CS, 0, wkv, 384, 0, bkv, nullptr, 0, 0,
              kv_buf, 384, 0, NTOK, 384, CS, 0, 1);
  launch_gemm(stream, s, CS, 0, wqp, 144, 0, bqp, nullptr, 0, 0,
              qp_raw, 144, 0, NTOK, 144, CS, 0, 1);
  launch_gemm(stream, s, CS, 0, wkvp, 432, 0, bkvp, nullptr, 0, 0,
              kvp_raw, 432, 0, NTOK, 432, CS, 0, 1);
  // pair bias: z (589824 x 128) @ wb (128 x 12)
  launch_gemm(stream, z, CZ, 0, wb, NH, 0, bbv, nullptr, 0, 0,
              bias_z, NH, 0, NTOK * NTOK, NH, CZ, 0, 1);
  // gate: sigmoid(s @ wg + bg)
  launch_gemm(stream, s, CS, 0, wg, CCAT, 0, bg, nullptr, 0, 0,
              g_buf, CCAT, 0, NTOK, CCAT, CS, 2, 1);

  // rigid transforms
  qp_transform<<<(NTOK * 48 + 255) / 256, 256, 0, stream>>>(qp_raw, rot, trans, q_pts);
  kvp_transform<<<(NTOK * 144 + 255) / 256, 256, 0, stream>>>(kvp_raw, rot, trans, k_pts, v_pts);

  // attention probs
  attn_scores_softmax<<<dim3(48, NH, 1), 256, 0, stream>>>(
      q_buf, kv_buf, q_pts, k_pts, bias_z, mask, hwp, a_buf);

  // o = a @ v           (batch over heads)
  launch_gemm(stream, a_buf, NH * NTOK, NTOK,
              kv_buf + CH, NH * 2 * CH, 2 * CH,
              nullptr, nullptr, 0, 0,
              fo, CCAT, CH, NTOK, CH, NTOK, 0, NH);
  // o_pt = a @ v_pts    (batch over heads, N=24 guarded)
  launch_gemm(stream, a_buf, NH * NTOK, NTOK,
              v_pts, NH * 24, 24,
              nullptr, nullptr, 0, 0,
              optb, NH * 24, 24, NTOK, 24, NTOK, 0, NH);
  // o_pair = a_i @ z_i  (batch over i, M=12 padded; TDM zero-fills rows 12..63)
  launch_gemm(stream, a_buf, NTOK, (long long)NH * NTOK,
              z, CZ, (long long)NTOK * CZ,
              nullptr, nullptr, 0, 0,
              fo + 576, CZ, CCAT, NH, CZ, NTOK, 0, NTOK);

  opt_finish<<<(NTOK * 96 + 255) / 256, 256, 0, stream>>>(optb, rot, trans, fo);
  mul_inplace<<<(NTOK * CCAT + 255) / 256, 256, 0, stream>>>(fo, g_buf, NTOK * CCAT);

  // s_ipa = s + (fo*g) @ wo + bo
  launch_gemm(stream, fo, CCAT, 0, wo, CS, 0, bo, s, CS, 0,
              s_ipa, CS, 0, NTOK, CS, CCAT, 0, 1);
  layernorm_row<<<NTOK, 256, 0, stream>>>(s_ipa, ln1g, ln1b, s_ln);

  // transition
  launch_gemm(stream, s_ln, CS, 0, tw1, CS, 0, tb1, nullptr, 0, 0,
              t1, CS, 0, NTOK, CS, CS, 1, 1);
  launch_gemm(stream, t1, CS, 0, tw2, CS, 0, tb2, nullptr, 0, 0,
              t2, CS, 0, NTOK, CS, CS, 1, 1);
  launch_gemm(stream, t2, CS, 0, tw3, CS, 0, tb3, s_ln, CS, 0,
              s2, CS, 0, NTOK, CS, CS, 0, 1);
  layernorm_row<<<NTOK, 256, 0, stream>>>(s2, ln2g, ln2b, out_s);

  backbone_update<<<(NTOK * 3 + 255) / 256, 256, 0, stream>>>(out_s, bbw, bbb, out_t);
}
